// Moduel2_48120813584665
// MI455X (gfx1250) — matrix operand order repaired and checked by reading, not compiled, not run on hardware
//
#include <hip/hip_runtime.h>
#include <stdint.h>

// LSTM for MI455X (gfx1250), two phases:
//  Phase 1 (k_xproj): xp[t][g][b] = X@W, full-device WMMA GEMM (bf16, f32 acc).
//    Removes the input projection (33% of serial-loop WMMAs) from the scan.
//  Phase 2 (k_lstm): 1024-step recurrence. 16 WGs (16-batch-row tiles),
//    16 wave32s each; wave w owns hidden cols [16w,16w+16) -> 4 gate tiles.
//    Per step per wave: 32x v_wmma_f32_16x16x32_bf16 (K=H=256), fragments
//    double-buffered against the L2 weight stream; gate preacts init from
//    xp (register-prefetched 1 step ahead, L2-prefetched 6 steps ahead via
//    global_prefetch_b8); cell state resident in VGPRs; h in LDS (16 KB,
//    double-buffered); one barrier per step.
// Weights ([W;U] transposed, bf16, 768 KB) stay L2-resident (192 MB L2).
// Workspace: xp 512 MiB + Xb 64 MiB + WUt 768 KiB (~577 MiB total).

#define B_  256
#define S_  1024
#define D_  128
#define H_  256
#define G_  1024   // 4H
#define KT_ 384    // D + H

typedef __attribute__((ext_vector_type(16))) __bf16         v16bf;
typedef __attribute__((ext_vector_type(8)))  float          v8f;
typedef __attribute__((ext_vector_type(8)))  unsigned short ush8;

union Frag16 { v16bf bf; ush8 u[2]; };

__device__ __forceinline__ unsigned short f2bf(float x) {
  unsigned u = __float_as_uint(x);
  u += 0x7FFFu + ((u >> 16) & 1u);          // round-to-nearest-even
  return (unsigned short)(u >> 16);
}
__device__ __forceinline__ float bf2f(unsigned short h) {
  return __uint_as_float(((unsigned)h) << 16);
}
__device__ __forceinline__ float sigf(float x) {
  return __builtin_amdgcn_rcpf(1.0f + __expf(-x));
}
__device__ __forceinline__ float tanhfast(float x) {
  float e = __expf(-2.0f * x);
  return (1.0f - e) * __builtin_amdgcn_rcpf(1.0f + e);
}

// ---- prep kernel 1: X fp32 -> bf16 (one HBM sweep) ----
__global__ void k_cvt_x(const float* __restrict__ X,
                        unsigned short* __restrict__ Xb, int n) {
  for (int i = blockIdx.x * blockDim.x + threadIdx.x; i < n;
       i += gridDim.x * blockDim.x)
    Xb[i] = f2bf(X[i]);
}

// ---- prep kernel 2: WUt[n][k] = (k<D ? W[k][n] : U[k-D][n]) as bf16 ----
__global__ void k_build_wut(const float* __restrict__ W,
                            const float* __restrict__ U,
                            unsigned short* __restrict__ WUt) {
  int i = blockIdx.x * blockDim.x + threadIdx.x;
  if (i >= G_ * KT_) return;
  int n = i / KT_, k = i % KT_;
  float v = (k < D_) ? W[(size_t)k * G_ + n] : U[(size_t)(k - D_) * G_ + n];
  WUt[i] = f2bf(v);
}

// ---- phase 1: xp[t][g][b] = sum_d Xb[b][t][d] * Wt[g][d]  (no bias) ----
// One wave per (t, 16-batch tile); A (X tile, K=128) in registers, loop 64
// g-tiles streaming B from L2, one b128 store per lane per tile.
__global__ __launch_bounds__(512) void k_xproj(
    const unsigned short* __restrict__ Xb,
    const unsigned short* __restrict__ WUt,
    unsigned short* __restrict__ xp) {
  const int wave  = threadIdx.x >> 5;
  const int lane  = threadIdx.x & 31;
  const int col16 = lane & 15;
  const int half  = lane >> 4;
  const int wid   = blockIdx.x * 16 + wave;   // 0..16383
  const int t     = wid >> 4;
  const int bbase = (wid & 15) * 16;

  Frag16 a[4];
  const unsigned short* ax = Xb + ((size_t)(bbase + col16) * S_ + t) * D_;
#pragma unroll
  for (int kc = 0; kc < 4; ++kc) {
    a[kc].u[0] = *(const ush8*)(ax + kc * 32 + half * 8);
    a[kc].u[1] = *(const ush8*)(ax + kc * 32 + 16 + half * 8);
  }
  for (int gt = 0; gt < 64; ++gt) {
    const unsigned short* bw = WUt + (size_t)(gt * 16 + col16) * KT_ + half * 8;
    v8f acc;
#pragma unroll
    for (int j = 0; j < 8; ++j) acc[j] = 0.0f;
#pragma unroll
    for (int kc = 0; kc < 4; ++kc) {
      Frag16 bf;
      bf.u[0] = *(const ush8*)(bw + kc * 32);
      bf.u[1] = *(const ush8*)(bw + kc * 32 + 16);
      acc = __builtin_amdgcn_wmma_f32_16x16x32_bf16(false, a[kc].bf, false, bf.bf,
                                                    (short)0, acc, false, false);
    }
    ush8 o;
#pragma unroll
    for (int j = 0; j < 8; ++j) o[j] = f2bf(acc[j]);
    // lane holds col g = gt*16+col16, rows b = bbase+half*8+j (contiguous)
    *(ush8*)&xp[((size_t)t * G_ + gt * 16 + col16) * B_ + bbase + half * 8] = o;
  }
}

// ---- phase 2: fused LSTM scan ----
__global__ __launch_bounds__(512, 1) void k_lstm(
    const unsigned short* __restrict__ xp,   // [S][G][B] bf16 preacts (no bias)
    const unsigned short* __restrict__ WUt,  // [G][KT] bf16; U part at k>=128
    const float* __restrict__ h0,            // [B][H]
    const float* __restrict__ bias2,         // [G]
    const float* __restrict__ fc2w,          // [H]
    const float* __restrict__ fc2b,          // [1]
    float* __restrict__ out)                 // [B][S]
{
  __shared__ __align__(16) unsigned short A[2][16][H_];  // h double-buffer

  const int tid   = threadIdx.x;
  const int wave  = tid >> 5;
  const int lane  = tid & 31;
  const int bbase = blockIdx.x * 16;
  const int col16 = lane & 15;
  const int half  = lane >> 4;

  const int   wcol = wave * 16 + col16;
  const float b_i  = bias2[wcol];
  const float b_f  = bias2[H_     + wcol];
  const float b_g  = bias2[2 * H_ + wcol];
  const float b_o  = bias2[3 * H_ + wcol];

  float fcw[8];
#pragma unroll
  for (int j = 0; j < 8; ++j) fcw[j] = fc2w[lane * 8 + j];
  const float fcb = fc2b[0];

  // h_0 into A[0]
  {
    int r  = tid >> 5;
    int c0 = (tid & 31) * 8;
#pragma unroll
    for (int j = 0; j < 8; ++j)
      A[0][r][c0 + j] = f2bf(h0[(size_t)(bbase + r) * H_ + c0 + j]);
  }
  __syncthreads();

  v8f cst;
#pragma unroll
  for (int j = 0; j < 8; ++j) cst[j] = 0.0f;

  // U-part B-fragment row bases (k >= D_ region of WUt).
  const unsigned short* brow[4];
  // xp per-tile element bases (step 0); advance by G_*B_ per step.
  const unsigned short* xprow[4];
#pragma unroll
  for (int tt = 0; tt < 4; ++tt) {
    brow[tt]  = WUt + (size_t)(tt * H_ + wave * 16 + col16) * KT_ + D_ + half * 8;
    xprow[tt] = xp + ((size_t)tt * H_ + wave * 16 + col16) * B_ + bbase + half * 8;
  }
  const unsigned short* arow0 = &A[0][lane & 15][0];
  const size_t step_elems = (size_t)G_ * B_;

  // Register-prefetch xp for t=0.
  ush8 xn[4];
#pragma unroll
  for (int tt = 0; tt < 4; ++tt) xn[tt] = *(const ush8*)(xprow[tt]);

  for (int t = 0; t < S_; ++t) {
    const int cur = t & 1, nxt = cur ^ 1;

    // L2 prefetch of the xp slab 6 steps ahead (global_prefetch_b8):
    // 2 instrs cover this wave's 64 row-segments (4 tiles x 16 g-rows).
    {
      int tp = t + 6; if (tp > S_ - 1) tp = S_ - 1;
#pragma unroll
      for (int pi = 0; pi < 2; ++pi) {
        int s2 = lane + pi * 32;
        int tt = s2 >> 4, gi = s2 & 15;
        const unsigned short* pa =
            xp + ((size_t)tp * G_ + tt * H_ + wave * 16 + gi) * B_ + bbase;
        __builtin_prefetch(pa, 0, 3);
      }
    }

    // Accumulators = xp preacts (prefetched last step) + bias.
    v8f ai, af, ag, ao;
#pragma unroll
    for (int j = 0; j < 8; ++j) {
      ai[j] = bf2f(xn[0][j]) + b_i;
      af[j] = bf2f(xn[1][j]) + b_f;
      ag[j] = bf2f(xn[2][j]) + b_g;
      ao[j] = bf2f(xn[3][j]) + b_o;
    }
    // Issue next step's xp loads now; they drain under the WMMA stream.
    {
      size_t off = (size_t)((t + 1 < S_) ? t + 1 : t) * step_elems;
#pragma unroll
      for (int tt = 0; tt < 4; ++tt)
        xn[tt] = *(const ush8*)(xprow[tt] + off);
    }

    // gates += h_t @ U : 32 WMMAs (8 K-chunks x 4 tiles), double-buffered.
    const unsigned short* arow = arow0 + (size_t)cur * 16 * H_;
    Frag16 a[2];
    Frag16 b[2][4];
    a[0].u[0] = *(const ush8*)(arow + half * 8);
    a[0].u[1] = *(const ush8*)(arow + 16 + half * 8);
#pragma unroll
    for (int tt = 0; tt < 4; ++tt) {
      b[0][tt].u[0] = *(const ush8*)(brow[tt]);
      b[0][tt].u[1] = *(const ush8*)(brow[tt] + 16);
    }
#pragma unroll
    for (int kc = 0; kc < 8; ++kc) {
      const int c = kc & 1, nb = c ^ 1;
      if (kc + 1 < 8) {
        const int kb = (kc + 1) * 32;
        a[nb].u[0] = *(const ush8*)(arow + kb + half * 8);
        a[nb].u[1] = *(const ush8*)(arow + kb + 16 + half * 8);
#pragma unroll
        for (int tt = 0; tt < 4; ++tt) {
          b[nb][tt].u[0] = *(const ush8*)(brow[tt] + kb);
          b[nb][tt].u[1] = *(const ush8*)(brow[tt] + kb + 16);
        }
      }
      ai = __builtin_amdgcn_wmma_f32_16x16x32_bf16(false, a[c].bf, false, b[c][0].bf,
                                                   (short)0, ai, false, false);
      af = __builtin_amdgcn_wmma_f32_16x16x32_bf16(false, a[c].bf, false, b[c][1].bf,
                                                   (short)0, af, false, false);
      ag = __builtin_amdgcn_wmma_f32_16x16x32_bf16(false, a[c].bf, false, b[c][2].bf,
                                                   (short)0, ag, false, false);
      ao = __builtin_amdgcn_wmma_f32_16x16x32_bf16(false, a[c].bf, false, b[c][3].bf,
                                                   (short)0, ao, false, false);
    }

    // Activations, cell update (fp32 in regs), write h_{t+1} to A[nxt].
#pragma unroll
    for (int j = 0; j < 8; ++j) {
      float iv = sigf(ai[j]);
      float fv = sigf(af[j]);
      float gv = tanhfast(ag[j]);
      float ov = sigf(ao[j]);
      float cv = fv * cst[j] + iv * gv;
      cst[j]   = cv;
      float hv = ov * tanhfast(cv);
      int   m  = j + half * 8;
      A[nxt][m][wcol] = f2bf(hv);
    }

    __syncthreads();   // single per-step barrier

    // y_t = h_new @ fc2w + fc2b : wave r reduces batch row r. (Safe without
    // trailing barrier: next write to A[nxt] is step t+2's h-store, behind
    // the step t+1 barrier.)
    {
      ush8 hvv = *(const ush8*)&A[nxt][wave][lane * 8];
      float p = 0.0f;
#pragma unroll
      for (int j = 0; j < 8; ++j) p += bf2f(hvv[j]) * fcw[j];
#pragma unroll
      for (int off = 16; off >= 1; off >>= 1) p += __shfl_xor(p, off, 32);
      if (lane == 0) out[(size_t)(bbase + wave) * S_ + t] = p + fcb;
    }
  }
}

extern "C" void kernel_launch(void* const* d_in, const int* in_sizes, int n_in,
                              void* d_out, int out_size, void* d_ws, size_t ws_size,
                              hipStream_t stream) {
  const float* X  = (const float*)d_in[0];
  const float* h0 = (const float*)d_in[1];
  const float* W  = (const float*)d_in[2];
  const float* U  = (const float*)d_in[3];
  const float* b2 = (const float*)d_in[4];
  const float* fw = (const float*)d_in[5];
  const float* fb = (const float*)d_in[6];
  float* out = (float*)d_out;

  // Workspace: xp [S][G][B] bf16 (512 MiB), Xb (64 MiB), WUt (768 KiB).
  unsigned short* xp  = (unsigned short*)d_ws;
  unsigned short* Xb  = (unsigned short*)((char*)d_ws + (size_t)S_ * G_ * B_ * 2);
  unsigned short* WUt = (unsigned short*)((char*)Xb + (size_t)B_ * S_ * D_ * 2);

  int nx = B_ * S_ * D_;
  k_cvt_x<<<4096, 256, 0, stream>>>(X, Xb, nx);
  k_build_wut<<<(G_ * KT_ + 255) / 256, 256, 0, stream>>>(W, U, WUt);
  k_xproj<<<1024, 512, 0, stream>>>(Xb, WUt, xp);
  k_lstm<<<B_ / 16, 512, 0, stream>>>(xp, WUt, h0, b2, fw, fb, out);
}
